// LSTM_encoder_82669530514117
// MI455X (gfx1250) — compile-verified, matrix operand order checked by reading, not run on hardware
//
#include <hip/hip_runtime.h>
#include <hip/hip_bf16.h>

typedef __attribute__((ext_vector_type(16))) __bf16 v16bf;
typedef __attribute__((ext_vector_type(8)))  float  v8f;

#define B_   256
#define T_   512
#define IN_  1024
#define H_   1024
#define KTOT (IN_ + H_)   // 2048
#define NG   4

#define BM 64
#define BN 32
#define BK 32

static __device__ __forceinline__ float sigmoidf_(float x) {
    return 1.0f / (1.0f + __expf(-x));
}

// Pack Wi|Wf|Wo|Wc (fp32, [2048,1024] row-major, rows = K) into one bf16
// matrix laid out [4096 cols][2048 K] (K contiguous per column) so B-matrix
// WMMA fragments are contiguous 16B loads.
__global__ void prep_weights_kernel(const float* __restrict__ Wi,
                                    const float* __restrict__ Wf,
                                    const float* __restrict__ Wo,
                                    const float* __restrict__ Wc,
                                    __bf16* __restrict__ wall_t) {
    size_t idx = (size_t)blockIdx.x * blockDim.x + threadIdx.x; // 4096*2048 total
    int k   = (int)(idx & (size_t)(KTOT - 1));
    int col = (int)(idx >> 11);
    int g   = col >> 10;
    int n   = col & (H_ - 1);
    const float* W = (g == 0) ? Wi : (g == 1) ? Wf : (g == 2) ? Wo : Wc;
    wall_t[idx] = (__bf16)W[(size_t)k * H_ + n];
}

__global__ void prep_bias_kernel(const float* __restrict__ bi,
                                 const float* __restrict__ bf,
                                 const float* __restrict__ bo,
                                 const float* __restrict__ bc,
                                 float* __restrict__ ball) {
    int i = blockIdx.x * blockDim.x + threadIdx.x; // 4096
    int g = i >> 10;
    int n = i & (H_ - 1);
    const float* b = (g == 0) ? bi : (g == 1) ? bf : (g == 2) ? bo : bc;
    ball[i] = b[n];
}

__global__ void init_state_kernel(__bf16* __restrict__ h_a,
                                  float* __restrict__ c_buf) {
    int i = blockIdx.x * blockDim.x + threadIdx.x; // 256*1024
    h_a[i]   = (__bf16)0.0f;
    c_buf[i] = 0.0f;
}

// One LSTM timestep. Each workgroup: [BM batch x BN hid] tile, all 4 gates.
// gates = [x_t | h] @ Wall + b ; c' = f*c + i*g ; h' = o*tanh(c')
// Software pipeline: double-buffered LDS, next K-block's global loads staged
// in registers and stored to LDS *after* the current WMMAs, one barrier/iter.
// All four B fragments are hoisted ahead of the WMMA sequence so the LDS
// loads pipeline instead of serializing each WMMA on its own ds_loads.
__global__ __launch_bounds__(256)
void lstm_step_kernel(const float*  __restrict__ x,       // [B, T, IN] fp32
                      const __bf16* __restrict__ wall_t,  // [4*H][KTOT] bf16
                      const float*  __restrict__ ball,    // [4*H]
                      const __bf16* __restrict__ h_in,    // [B, H] bf16
                      __bf16*       __restrict__ h_out,   // [B, H] bf16
                      float*        __restrict__ h_f32,   // [B, H] fp32 (d_out h-half)
                      float*        __restrict__ c_buf,   // [B, H] fp32 (d_out c-half)
                      int t) {
    __shared__ __bf16 lA[2][BM][BK];        // 2 x 4 KB
    __shared__ __bf16 lB[2][NG][BN][BK];    // 2 x 8 KB

    const int tid  = threadIdx.x;
    const int lane = tid & 31;           // wave32
    const int wave = tid >> 5;           // 8 waves
    const int wm   = wave & 3;           // M sub-tile 0..3
    const int wn   = wave >> 2;          // N sub-tile 0..1
    const int b0   = blockIdx.y * BM;
    const int j0   = blockIdx.x * BN;

    // A-tile cooperative load: 4 threads per row, 8 K each
    const int arow = tid >> 2;
    const int aseg = (tid & 3) * 8;
    // B-tile cooperative load: 2 threads per column, 16 K each
    const int bcol = tid >> 1;           // 0..127 -> gate*32 + n
    const int bg   = bcol >> 5;
    const int bnn  = bcol & 31;
    const int bks  = (tid & 1) * 16;
    const size_t bgcol = (size_t)(bg * H_ + j0 + bnn);

    const int half = lane >> 4;          // half-wave -> K group
    const int l16  = lane & 15;

    const float*  xrow = x + ((size_t)(b0 + arow) * T_ + t) * IN_;
    const __bf16* hrow = h_in + (size_t)(b0 + arow) * H_;
    const __bf16* wcol = wall_t + bgcol * KTOT + bks;

    v8f acc[NG];
    {
        v8f z = {0.f, 0.f, 0.f, 0.f, 0.f, 0.f, 0.f, 0.f};
        #pragma unroll
        for (int g = 0; g < NG; ++g) acc[g] = z;
    }

    // ---- preload K-block 0 into buffer 0 (always x region) ----
    {
        const float* sx = xrow + aseg;
        float4 f0 = ((const float4*)sx)[0];
        float4 f1 = ((const float4*)sx)[1];
        __bf16* dA = &lA[0][arow][aseg];
        dA[0] = (__bf16)f0.x; dA[1] = (__bf16)f0.y;
        dA[2] = (__bf16)f0.z; dA[3] = (__bf16)f0.w;
        dA[4] = (__bf16)f1.x; dA[5] = (__bf16)f1.y;
        dA[6] = (__bf16)f1.z; dA[7] = (__bf16)f1.w;
        uint4 w0 = ((const uint4*)wcol)[0];
        uint4 w1 = ((const uint4*)wcol)[1];
        uint4* dB = (uint4*)&lB[0][bg][bnn][bks];
        dB[0] = w0; dB[1] = w1;
    }
    __syncthreads();

    int buf = 0;
    for (int kb = 0; kb < KTOT; kb += BK, buf ^= 1) {
        const int  kn        = kb + BK;
        const bool have_next = (kn < KTOT);
        const bool next_is_x = (kn < IN_);

        // ---- (a) issue next K-block's global loads into registers ----
        float4 xf0 = {}, xf1 = {};
        uint4  hu  = {};
        uint4  w0  = {}, w1 = {};
        if (have_next) {
            if (next_is_x) {
                const float* sx = xrow + kn + aseg;
                xf0 = ((const float4*)sx)[0];
                xf1 = ((const float4*)sx)[1];
            } else {
                hu = *(const uint4*)(hrow + (kn - IN_) + aseg);
            }
            const __bf16* sw = wcol + kn;
            w0 = ((const uint4*)sw)[0];
            w1 = ((const uint4*)sw)[1];
            __builtin_prefetch((const void*)(sw + 4 * BK), 0, 3); // global_prefetch_b8, near
        }

        // ---- (b) WMMA on current buffer: hoist A + all 4 B fragments ----
        union Frag { v16bf v; uint4 u[2]; };
        Frag a, bfr[NG];
        const int mr = wm * 16 + l16;
        a.u[0] = *(const uint4*)&lA[buf][mr][half * 8];       // K {0..7} / {8..15}
        a.u[1] = *(const uint4*)&lA[buf][mr][16 + half * 8];  // K {16..23} / {24..31}
        const int nr = wn * 16 + l16;
        #pragma unroll
        for (int g = 0; g < NG; ++g) {
            bfr[g].u[0] = *(const uint4*)&lB[buf][g][nr][half * 8];
            bfr[g].u[1] = *(const uint4*)&lB[buf][g][nr][16 + half * 8];
        }
        #pragma unroll
        for (int g = 0; g < NG; ++g) {
            acc[g] = __builtin_amdgcn_wmma_f32_16x16x32_bf16(
                false, a.v, false, bfr[g].v, (short)0, acc[g], false, false);
        }

        // ---- (c) store staged registers into the other buffer ----
        if (have_next) {
            const int nb = buf ^ 1;
            if (next_is_x) {
                __bf16* dA = &lA[nb][arow][aseg];
                dA[0] = (__bf16)xf0.x; dA[1] = (__bf16)xf0.y;
                dA[2] = (__bf16)xf0.z; dA[3] = (__bf16)xf0.w;
                dA[4] = (__bf16)xf1.x; dA[5] = (__bf16)xf1.y;
                dA[6] = (__bf16)xf1.z; dA[7] = (__bf16)xf1.w;
            } else {
                *(uint4*)&lA[nb][arow][aseg] = hu;
            }
            uint4* dB = (uint4*)&lB[nb][bg][bnn][bks];
            dB[0] = w0; dB[1] = w1;
        }
        __syncthreads();
    }

    // ---- epilogue: gate nonlinearity + state update ----
    // C/D layout: lane l, vgpr r -> M = (l>>4)*8 + r, N = l&15
    const int j  = j0 + wn * 16 + l16;
    const float bi_v = ball[j];
    const float bf_v = ball[H_ + j];
    const float bo_v = ball[2 * H_ + j];
    const float bc_v = ball[3 * H_ + j];

    #pragma unroll
    for (int r = 0; r < 8; ++r) {
        const int bidx = b0 + wm * 16 + half * 8 + r;
        const size_t off = (size_t)bidx * H_ + j;
        float iv = sigmoidf_(acc[0][r] + bi_v);
        float fv = sigmoidf_(acc[1][r] + bf_v);
        float ov = sigmoidf_(acc[2][r] + bo_v);
        float gv = tanhf(acc[3][r] + bc_v);
        float cn = fv * c_buf[off] + iv * gv;
        float hn = ov * tanhf(cn);
        c_buf[off] = cn;
        h_f32[off] = hn;
        h_out[off] = (__bf16)hn;
    }
}

extern "C" void kernel_launch(void* const* d_in, const int* in_sizes, int n_in,
                              void* d_out, int out_size, void* d_ws, size_t ws_size,
                              hipStream_t stream) {
    const float* x  = (const float*)d_in[0];
    const float* Wf = (const float*)d_in[1];
    const float* bf = (const float*)d_in[2];
    const float* Wi = (const float*)d_in[3];
    const float* bi = (const float*)d_in[4];
    const float* Wo = (const float*)d_in[5];
    const float* bo = (const float*)d_in[6];
    const float* Wc = (const float*)d_in[7];
    const float* bc = (const float*)d_in[8];

    // Workspace carve-up (~17.8 MB): packed bf16 weights + bias + h ping-pong
    char* p = (char*)d_ws;
    __bf16* wall_t = (__bf16*)p;  p += (size_t)NG * H_ * KTOT * sizeof(__bf16); // 16 MB
    float*  ball   = (float*)p;   p += (size_t)NG * H_ * sizeof(float);         // 16 KB
    __bf16* h_a    = (__bf16*)p;  p += (size_t)B_ * H_ * sizeof(__bf16);        // 512 KB
    __bf16* h_b    = (__bf16*)p;

    float* h_f32 = (float*)d_out;                   // [B, H] final h
    float* c_buf = (float*)d_out + (size_t)B_ * H_; // [B, H] c, updated in place

    prep_weights_kernel<<<(NG * H_ * KTOT) / 256, 256, 0, stream>>>(Wi, Wf, Wo, Wc, wall_t);
    prep_bias_kernel<<<(NG * H_) / 256, 256, 0, stream>>>(bi, bf, bo, bc, ball);
    init_state_kernel<<<(B_ * H_) / 256, 256, 0, stream>>>(h_a, c_buf);

    for (int t = 0; t < T_; ++t) {
        const __bf16* hin = (t & 1) ? h_b : h_a;
        __bf16*       hnx = (t & 1) ? h_a : h_b;
        lstm_step_kernel<<<dim3(H_ / BN, B_ / BM), 256, 0, stream>>>(
            x, wall_t, ball, hin, hnx, h_f32, c_buf, t);
    }
}
